// GRUCell_28003186770568
// MI455X (gfx1250) — matrix operand order repaired and checked by reading, not compiled, not run on hardware
//
#include <hip/hip_runtime.h>
#include <hip/hip_bf16.h>

// ---------------------------------------------------------------------------
// GRU over SEQ=4096 steps, I=H=2048.
//   Phase 1: Gx_g = X @ Wg[:I] + b_g (all steps/gates) -> bf16 WMMA GEMM with
//            LDS-tiled B, double-buffered via async global->LDS copies.
//   Phase 2: persistent sequential kernel, grid barrier, bf16 h-part weights
//            (24 MB, L2-resident) with fp32 accumulation.
// ---------------------------------------------------------------------------

#define SEQ   4096
#define ISZ   2048
#define HID   2048

// sequential kernel config
#define SNBLK 64
#define TPB   256
#define CPB   32
#define CHUNKS 8
#define KSPAN (HID / CHUNKS)

// GEMM tile config
#define GM_MBLK 128          // M rows per block (8 waves x 16)
#define GM_NBLK 64           // N cols per block
#define GM_KSLAB 64          // K per LDS slab
#define GM_BPAD 8            // pad (ushorts) -> 144B row stride, conflict-free
#define GM_BSTRIDE (GM_KSLAB + GM_BPAD)

typedef __attribute__((ext_vector_type(16))) __bf16 v16bf;
typedef __attribute__((ext_vector_type(8)))  float  v8f;

union Frag { v16bf v; uint4 q[2]; };

__device__ __forceinline__ unsigned short f2bf(float x) {
  unsigned int u = __float_as_uint(x);
  unsigned int r = (u + 0x7FFFu + ((u >> 16) & 1u)) >> 16;   // RNE
  return (unsigned short)r;
}
__device__ __forceinline__ float bf2f(unsigned short b) {
  return __uint_as_float((unsigned int)b << 16);
}

// async global -> LDS copy, 16 bytes per lane (ASYNCcnt-tracked)
#define ASYNC_B128(ldsoff, gptr)                                   \
  asm volatile("global_load_async_to_lds_b128 %0, %1, off"         \
               :: "v"(ldsoff), "v"(gptr) : "memory")

// ------------------------------- prep kernels ------------------------------

__global__ void k_init(float* h_cur, float* rh, unsigned int* bar) {
  int t = threadIdx.x;
  for (int i = t; i < HID; i += TPB) { h_cur[i] = 0.f; rh[i] = 0.f; }
  if (t < 8) bar[t] = 0u;
}

__global__ void k_cvt_x(const float* __restrict__ X, unsigned short* __restrict__ Xb) {
  size_t n = (size_t)SEQ * ISZ;
  for (size_t i = (size_t)blockIdx.x * blockDim.x + threadIdx.x; i < n;
       i += (size_t)gridDim.x * blockDim.x)
    Xb[i] = f2bf(X[i]);
}

// x-part, transposed to n-major: Wt[n][k] = bf16(W[k][n]), k < ISZ
__global__ void k_cvt_wt(const float* __restrict__ W, unsigned short* __restrict__ Wt) {
  int k = blockIdx.x * blockDim.x + threadIdx.x;
  int n = blockIdx.y;
  Wt[(size_t)n * ISZ + k] = f2bf(W[(size_t)k * HID + n]);
}

// h-part, layout preserved: Wh[k][n] = bf16(W[ISZ+k][n])
__global__ void k_cvt_wh(const float* __restrict__ W, unsigned short* __restrict__ Wh) {
  size_t nel = (size_t)HID * HID;
  for (size_t i = (size_t)blockIdx.x * blockDim.x + threadIdx.x; i < nel;
       i += (size_t)gridDim.x * blockDim.x)
    Wh[i] = f2bf(W[(size_t)ISZ * HID + i]);
}

// ------------------------- phase 1: bf16 WMMA GEMM -------------------------
// Block = 128(M) x 64(N); 8 waves each own a 16x64 strip, sharing the B slab
// in LDS. B slabs (64n x 64k bf16) are double-buffered and filled with
// global_load_async_to_lds_b128 (2 per thread per slab), drained with
// s_wait_asynccnt. A fragments come straight from global; A and B share contiguous k order:
//   A 16x32 bf16: half 0 -> K kk..kk+15, half 1 -> K kk+16..kk+31.
//   B 32x16 bf16: half 0 -> K kk..kk+15, half 1 -> K kk+16..kk+31.
__global__ void __launch_bounds__(256)
k_gemm_x(const unsigned short* __restrict__ Xb,    // [SEQ][ISZ]
         const unsigned short* __restrict__ Wt,    // [HID][ISZ] n-major
         const float* __restrict__ bias,           // [HID]
         float* __restrict__ Gx)                   // [SEQ][HID]
{
  __shared__ __align__(16) unsigned short Bs[2][GM_NBLK * GM_BSTRIDE];

  const int tid  = threadIdx.x;
  const int wv   = tid >> 5;
  const int lane = tid & 31;
  const int half = lane >> 4;
  const int lm   = lane & 15;

  const int mblock = (int)blockIdx.x >> 5;             // 32 m-blocks
  const int nblock = (int)blockIdx.x & 31;             // 32 n-blocks
  const int mbase  = mblock * GM_MBLK + wv * 16;
  const int nbase  = nblock * GM_NBLK;

  // B-copy job: thread covers (row r0, 16B chunk c0) and (row r0+32, c0)
  const int r0 = tid >> 3;                             // 0..31
  const int c0 = tid & 7;                              // 0..7 (x8 ushorts)
  const unsigned short* gB0 = Wt + (size_t)(nbase + r0) * ISZ + c0 * 8;
  const unsigned short* gB1 = gB0 + (size_t)32 * ISZ;
  const unsigned lbs   = (unsigned)(GM_NBLK * GM_BSTRIDE * sizeof(unsigned short));
  const unsigned l0    = (unsigned)(uintptr_t)&Bs[0][r0 * GM_BSTRIDE + c0 * 8];
  const unsigned l1    = (unsigned)(uintptr_t)&Bs[0][(r0 + 32) * GM_BSTRIDE + c0 * 8];

  v8f acc[4];
#pragma unroll
  for (int t = 0; t < 4; ++t) {
    float bv = bias[nbase + t * 16 + lm];
#pragma unroll
    for (int v = 0; v < 8; ++v) acc[t][v] = bv;
  }

  const unsigned short* arow = Xb + (size_t)(mbase + lm) * ISZ;

  const int NS = ISZ / GM_KSLAB;                       // 32 slabs
  // preload slab 0 into buffer 0
  ASYNC_B128(l0, gB0);
  ASYNC_B128(l1, gB1);

  for (int s = 0; s < NS; ++s) {
    const int buf = s & 1;
    if (s + 1 < NS) {
      const unsigned boff = (unsigned)((s + 1) & 1) * lbs;
      const size_t gk = (size_t)(s + 1) * GM_KSLAB;
      ASYNC_B128(l0 + boff, gB0 + gk);
      ASYNC_B128(l1 + boff, gB1 + gk);
      __builtin_prefetch(arow + gk, 0, 0);             // global_prefetch_b8
      asm volatile("s_wait_asynccnt 0x2" ::: "memory"); // slab s landed
    } else {
      asm volatile("s_wait_asynccnt 0x0" ::: "memory");
    }
    __syncthreads();

    const unsigned short* bsb = &Bs[buf][0];
    const int ks = s * GM_KSLAB;
#pragma unroll
    for (int kk = 0; kk < GM_KSLAB; kk += 32) {
      Frag a;
      a.q[0] = *(const uint4*)(arow + ks + kk + half * 16);
      a.q[1] = *(const uint4*)(arow + ks + kk + half * 16 + 8);
#pragma unroll
      for (int t = 0; t < 4; ++t) {
        const unsigned short* lp =
            bsb + (t * 16 + lm) * GM_BSTRIDE + kk + half * 16;
        Frag b;
        b.q[0] = *(const uint4*)(lp);                  // ds_load_b128
        b.q[1] = *(const uint4*)(lp + 8);
        acc[t] = __builtin_amdgcn_wmma_f32_16x16x32_bf16(
            false, a.v, false, b.v, (short)0, acc[t], false, false);
      }
    }
    __syncthreads();                                   // before slab s+2 overwrite
  }

#pragma unroll
  for (int t = 0; t < 4; ++t) {
    float* o = Gx + (size_t)(mbase + 8 * half) * HID + nbase + t * 16 + lm;
#pragma unroll
    for (int v = 0; v < 8; ++v) o[(size_t)v * HID] = acc[t][v];   // M = v + 8*half
  }
}

// ----------------------- phase 2: sequential recurrence --------------------

__device__ __forceinline__ void grid_barrier(unsigned int* cnt, unsigned int* gen) {
  __syncthreads();
  if (threadIdx.x == 0) {
    unsigned int g = __hip_atomic_load(gen, __ATOMIC_ACQUIRE, __HIP_MEMORY_SCOPE_AGENT);
    unsigned int a = __hip_atomic_fetch_add(cnt, 1u, __ATOMIC_ACQ_REL, __HIP_MEMORY_SCOPE_AGENT);
    if (a == SNBLK - 1) {
      __hip_atomic_store(cnt, 0u, __ATOMIC_RELAXED, __HIP_MEMORY_SCOPE_AGENT);
      __hip_atomic_fetch_add(gen, 1u, __ATOMIC_ACQ_REL, __HIP_MEMORY_SCOPE_AGENT);
    } else {
      while (__hip_atomic_load(gen, __ATOMIC_ACQUIRE, __HIP_MEMORY_SCOPE_AGENT) == g)
        __builtin_amdgcn_s_sleep(2);
    }
  }
  __syncthreads();
}

__global__ void __launch_bounds__(TPB)
k_gru_seq(const unsigned short* __restrict__ Whr,   // [HID][HID] bf16 h-part
          const unsigned short* __restrict__ Whu,
          const unsigned short* __restrict__ Whc,
          const float* __restrict__ Gxr, const float* __restrict__ Gxu,
          const float* __restrict__ Gxc,
          float* __restrict__ h_cur, float* __restrict__ rh,
          unsigned int* bar, float* __restrict__ out)
{
  __shared__ float h_s[HID];
  __shared__ float rh_s[HID];
  __shared__ float pR[CHUNKS][CPB];
  __shared__ float pU[CHUNKS][CPB];
  __shared__ float pC[CHUNKS][CPB];
  __shared__ float u_s[CPB];
  __shared__ float ho_s[CPB];

  const int tid   = threadIdx.x;
  const int col   = tid & (CPB - 1);
  const int chunk = tid >> 5;
  const int n     = blockIdx.x * CPB + col;
  const int k0    = chunk * KSPAN;

  const unsigned short* wr = Whr + (size_t)k0 * HID + n;
  const unsigned short* wu = Whu + (size_t)k0 * HID + n;
  const unsigned short* wc = Whc + (size_t)k0 * HID + n;

  for (int s = 0; s < SEQ; ++s) {
    for (int i = tid; i < HID; i += TPB) h_s[i] = h_cur[i];
    __syncthreads();

    float aR = 0.f, aU = 0.f;
    const unsigned short* pr = wr;
    const unsigned short* pu = wu;
#pragma unroll 8
    for (int k = 0; k < KSPAN; ++k) {
      float hv = h_s[k0 + k];
      aR = fmaf(hv, bf2f(*pr), aR);
      aU = fmaf(hv, bf2f(*pu), aU);
      pr += HID; pu += HID;
    }
    pR[chunk][col] = aR;
    pU[chunk][col] = aU;
    __syncthreads();

    if (tid < CPB) {
      float sR = 0.f, sU = 0.f;
#pragma unroll
      for (int c = 0; c < CHUNKS; ++c) { sR += pR[c][col]; sU += pU[c][col]; }
      float r = 1.f / (1.f + __expf(-(Gxr[(size_t)s * HID + n] + sR)));
      float u = 1.f / (1.f + __expf(-(Gxu[(size_t)s * HID + n] + sU)));
      float hold = h_s[n];
      u_s[col]  = u;
      ho_s[col] = hold;
      rh[n] = r * hold;
    }
    __threadfence();
    grid_barrier(bar, bar + 1);

    for (int i = tid; i < HID; i += TPB) rh_s[i] = rh[i];
    __syncthreads();

    float aC = 0.f;
    const unsigned short* pc = wc;
#pragma unroll 8
    for (int k = 0; k < KSPAN; ++k) {
      aC = fmaf(rh_s[k0 + k], bf2f(*pc), aC);
      pc += HID;
    }
    pC[chunk][col] = aC;
    __syncthreads();

    if (tid < CPB) {
      float sC = 0.f;
#pragma unroll
      for (int c = 0; c < CHUNKS; ++c) sC += pC[c][col];
      float ht = tanhf(Gxc[(size_t)s * HID + n] + sC);
      float u  = u_s[col];
      float hn = ho_s[col] * u + (1.f - u) * ht;
      h_cur[n] = hn;
      out[(size_t)s * HID + n] = hn;
    }
    __threadfence();
    grid_barrier(bar, bar + 1);
  }
}

// --------------------------------- launch ----------------------------------

extern "C" void kernel_launch(void* const* d_in, const int* in_sizes, int n_in,
                              void* d_out, int out_size, void* d_ws, size_t ws_size,
                              hipStream_t stream) {
  (void)in_sizes; (void)n_in; (void)out_size; (void)ws_size;
  const float* X   = (const float*)d_in[0];
  const float* Wu  = (const float*)d_in[1];
  const float* bu  = (const float*)d_in[2];
  const float* Wr  = (const float*)d_in[3];
  const float* br  = (const float*)d_in[4];
  const float* Wc  = (const float*)d_in[5];
  const float* bc  = (const float*)d_in[6];
  float* out = (float*)d_out;

  char* ws = (char*)d_ws;
  size_t offXb  = 0;                                       // 16 MB
  size_t offWt  = offXb + (size_t)SEQ * ISZ * 2;           // 3 x 8 MB
  size_t offWh  = offWt + 3 * (size_t)HID * ISZ * 2;       // 3 x 8 MB
  size_t offGx  = offWh + 3 * (size_t)HID * HID * 2;       // 3 x 32 MB
  size_t offH   = offGx + 3 * (size_t)SEQ * HID * 4;
  size_t offRH  = offH  + (size_t)HID * 4;
  size_t offBar = offRH + (size_t)HID * 4;

  unsigned short* Xb  = (unsigned short*)(ws + offXb);
  unsigned short* WtR = (unsigned short*)(ws + offWt);
  unsigned short* WtU = WtR + (size_t)HID * ISZ;
  unsigned short* WtC = WtU + (size_t)HID * ISZ;
  unsigned short* WhR = (unsigned short*)(ws + offWh);
  unsigned short* WhU = WhR + (size_t)HID * HID;
  unsigned short* WhC = WhU + (size_t)HID * HID;
  float* GxR = (float*)(ws + offGx);
  float* GxU = GxR + (size_t)SEQ * HID;
  float* GxC = GxU + (size_t)SEQ * HID;
  float* hcur = (float*)(ws + offH);
  float* rh   = (float*)(ws + offRH);
  unsigned int* bar = (unsigned int*)(ws + offBar);

  k_init<<<1, TPB, 0, stream>>>(hcur, rh, bar);

  k_cvt_x<<<4096, 256, 0, stream>>>(X, Xb);
  dim3 wtg(ISZ / 256, HID);
  k_cvt_wt<<<wtg, 256, 0, stream>>>(Wr, WtR);
  k_cvt_wt<<<wtg, 256, 0, stream>>>(Wu, WtU);
  k_cvt_wt<<<wtg, 256, 0, stream>>>(Wc, WtC);
  k_cvt_wh<<<2048, 256, 0, stream>>>(Wr, WhR);
  k_cvt_wh<<<2048, 256, 0, stream>>>(Wu, WhU);
  k_cvt_wh<<<2048, 256, 0, stream>>>(Wc, WhC);

  // phase 1: 32 m-blocks x 32 n-blocks
  const int gemmBlocks = (SEQ / GM_MBLK) * (HID / GM_NBLK);  // 1024
  k_gemm_x<<<gemmBlocks, 256, 0, stream>>>(Xb, WtR, br, GxR);
  k_gemm_x<<<gemmBlocks, 256, 0, stream>>>(Xb, WtU, bu, GxU);
  k_gemm_x<<<gemmBlocks, 256, 0, stream>>>(Xb, WtC, bc, GxC);

  // phase 2: persistent sequential recurrence
  k_gru_seq<<<SNBLK, TPB, 0, stream>>>(WhR, WhU, WhC, GxR, GxU, GxC,
                                       hcur, rh, bar, out);
}
